// GraphAttentionLayer_48387101556937
// MI455X (gfx1250) — matrix operand order repaired and checked by reading, not compiled, not run on hardware
//
#include <hip/hip_runtime.h>

typedef __attribute__((ext_vector_type(16))) _Float16 v16h;
typedef __attribute__((ext_vector_type(8)))  float    v8f;

#define GAT_B   2
#define GAT_N   4096
#define GAT_INF 256
#define GAT_H   4
#define GAT_F   64
#define GAT_HF  256

__device__ __forceinline__ float gat_lrelu(float x) { return x > 0.f ? x : 0.2f * x; }

// ---------------------------------------------------------------------------
// Kernel 0: split W into hi/lo f16 pair, stored in WMMA B-operand swizzled
// layout: for (o-tile, k-chunk) lane L holds col o%16==L%16, element s holds
// k = 8*(L/16)+s (s<8) or 16+8*(L/16)+(s-8).  B[k][o] = W[o][k]  (h = x @ W^T).
// ---------------------------------------------------------------------------
__global__ void __launch_bounds__(256) k0_wprep(const float* __restrict__ W,
                                                _Float16* __restrict__ WhiB,
                                                _Float16* __restrict__ WloB) {
  const int t = blockIdx.x * 256 + threadIdx.x;   // over 256*256
  const int o = t >> 8, k = t & 255;
  const float v = W[(size_t)o * GAT_INF + k];
  const _Float16 hv = (_Float16)v;
  const _Float16 lv = (_Float16)(v - (float)hv);
  const int ot = o >> 4, c = o & 15;
  const int kc = k >> 5, kk = k & 31;
  const int L = c + ((kk & 8) << 1);
  const size_t idx = ((size_t)(ot * 8 + kc) * 32 + L) * 16 + ((kk & 7) + ((kk & 16) >> 1));
  WhiB[idx] = hv;
  WloB[idx] = lv;
}

// ---------------------------------------------------------------------------
// Kernel 1: h = x @ W^T on the matrix pipe, split-f16 (hi*hi + hi*lo + lo*hi)
// with f32 accumulation -> ~f32 precision.  Each wave: 16 rows x 64 cols.
// Also emits the f16 copy of h in the k2 B-operand swizzled layout.
// ---------------------------------------------------------------------------
__global__ void __launch_bounds__(256) k1_wmma(const float* __restrict__ x,
                                               const _Float16* __restrict__ WhiB,
                                               const _Float16* __restrict__ WloB,
                                               float* __restrict__ h,
                                               _Float16* __restrict__ hB) {
  const int wave = threadIdx.x >> 5;
  const int lane = threadIdx.x & 31;
  const int rt   = blockIdx.x * 2 + (wave >> 2);   // row tile over [B*N]/16
  const int og   = (wave & 3) * 64;                // col group
  const int r    = lane & 15;
  const int hig  = lane >> 4;
  const int koffA = hig * 8;
  const int i0   = rt * 16;

  const float* xrow = x + (size_t)(i0 + r) * GAT_INF;
  v8f acc[4] = {v8f{}, v8f{}, v8f{}, v8f{}};

  for (int kc = 0; kc < GAT_INF / 32; ++kc) {
    const int k0 = kc * 32;
    const float4* xq = (const float4*)(xrow + k0 + koffA);
    float xb[16];
    ((float4*)xb)[0] = xq[0]; ((float4*)xb)[1] = xq[1];
    ((float4*)xb)[2] = xq[4]; ((float4*)xb)[3] = xq[5];
    v16h ah, al;
#pragma unroll
    for (int j = 0; j < 16; ++j) {
      const float v = xb[j];
      const _Float16 hv = (_Float16)v;
      ah[j] = hv;
      al[j] = (_Float16)(v - (float)hv);
    }
#pragma unroll
    for (int t = 0; t < 4; ++t) {
      const size_t bi = (size_t)(((og >> 4) + t) * 8 + kc) * 32 + lane;
      const v16h bh = ((const v16h*)WhiB)[bi];
      const v16h bl = ((const v16h*)WloB)[bi];
      acc[t] = __builtin_amdgcn_wmma_f32_16x16x32_f16(false, ah, false, bh, (short)0, acc[t], false, false);
      acc[t] = __builtin_amdgcn_wmma_f32_16x16x32_f16(false, ah, false, bl, (short)0, acc[t], false, false);
      acc[t] = __builtin_amdgcn_wmma_f32_16x16x32_f16(false, al, false, bh, (short)0, acc[t], false, false);
    }
  }

  // epilogue: write h (f32) and swizzled f16 copy (k2 B-operand layout)
  const int rowSel = hig * 8;
#pragma unroll
  for (int t = 0; t < 4; ++t) {
#pragma unroll
    for (int j = 0; j < 8; ++j) {
      const int row = rowSel + j;
      const int i = i0 + row;
      const int o = og + t * 16 + r;
      const float val = acc[t][j];
      h[(size_t)i * GAT_HF + o] = val;
      const int b  = i >> 12;              // / N
      const int n  = i & (GAT_N - 1);
      const int hh = o >> 6;
      const int ft = (o >> 4) & 3;
      const int mc = n >> 5, kk = n & 31;
      const int L  = r + ((kk & 8) << 1);
      const size_t idx =
          ((((size_t)(b * GAT_H + hh) * 4 + ft) * (GAT_N / 32) + mc) * 32 + L) * 16 + ((kk & 7) + ((kk & 16) >> 1));
      hB[idx] = (_Float16)val;
    }
  }
}

// ---------------------------------------------------------------------------
// Kernel 1b: attention scores s_src/s_dst, stored transposed [B,H,N]
// ---------------------------------------------------------------------------
__global__ void __launch_bounds__(256) k1b_scores(const float* __restrict__ h,
                                                  const float* __restrict__ aSrc,
                                                  const float* __restrict__ aDst,
                                                  float* __restrict__ sSrcT,
                                                  float* __restrict__ sDstT) {
  const int t  = blockIdx.x * blockDim.x + threadIdx.x;  // over B*N*H
  const int hh = t & 3;
  const int n  = (t >> 2) & (GAT_N - 1);
  const int b  = t >> 14;
  const float* hp = h + (size_t)(b * GAT_N + n) * GAT_HF + hh * GAT_F;
  const float* as = aSrc + hh * GAT_F;
  const float* ad = aDst + hh * GAT_F;
  float ss = 0.f, sd = 0.f;
#pragma unroll 8
  for (int f = 0; f < GAT_F; ++f) {
    float v = hp[f];
    ss += v * as[f];
    sd += v * ad[f];
  }
  sSrcT[(size_t)(b * GAT_H + hh) * GAT_N + n] = ss;
  sDstT[(size_t)(b * GAT_H + hh) * GAT_N + n] = sd;
}

// ---------------------------------------------------------------------------
// Kernel 1c: Dmax[b,h] = max_n s_dst  (upper bound for the softmax shift)
// ---------------------------------------------------------------------------
__global__ void __launch_bounds__(256) k1c_dmax(const float* __restrict__ sDstT,
                                                float* __restrict__ dmax) {
  __shared__ float red[256];
  const float* p = sDstT + (size_t)blockIdx.x * GAT_N;
  float m = -3.0e38f;
  for (int i = threadIdx.x; i < GAT_N; i += 256) m = fmaxf(m, p[i]);
  red[threadIdx.x] = m;
  __syncthreads();
  for (int s = 128; s > 0; s >>= 1) {
    if (threadIdx.x < s) red[threadIdx.x] = fmaxf(red[threadIdx.x], red[threadIdx.x + s]);
    __syncthreads();
  }
  if (threadIdx.x == 0) dmax[blockIdx.x] = red[0];
}

// ---------------------------------------------------------------------------
// Kernel 2: fused masked-softmax attention + h' = attn @ h via WMMA.
// Block = 128 threads (4 waves = 4 heads) sharing a 32-row tile.  Each wave
// owns TWO 16-row A-tiles so every B (h) tile is used by 2 WMMAs -> halves
// the dominant L2 stream.  s_dst quads shared across both tiles.  Single
// pass: shift = LR(c_n + max_m d_m) is a valid row upper bound (LR monotone).
// ---------------------------------------------------------------------------
__global__ void __launch_bounds__(128) k2_attn(const float* __restrict__ adj,
                                               const float* __restrict__ sSrcT,
                                               const float* __restrict__ sDstT,
                                               const float* __restrict__ dmax,
                                               const _Float16* __restrict__ hB,
                                               float* __restrict__ out) {
  const int lane = threadIdx.x & 31;
  const int hh   = threadIdx.x >> 5;
  const int b    = blockIdx.y;
  const int n0   = blockIdx.x * 32;
  const int r    = lane & 15;          // row within tile (A/C layout)
  const int koff = (lane >> 4) << 3;   // K base: 0 or 8

  const int bh    = b * GAT_H + hh;
  const float dm  = dmax[bh];
  const float c0  = sSrcT[(size_t)bh * GAT_N + n0 + r];
  const float c1  = sSrcT[(size_t)bh * GAT_N + n0 + 16 + r];
  const float sh0 = gat_lrelu(c0 + dm);
  const float sh1 = gat_lrelu(c1 + dm);

  const float* adjR0 = adj + (size_t)(n0 + r) * GAT_N;
  const float* adjR1 = adj + (size_t)(n0 + 16 + r) * GAT_N;
  const float* dRow  = sDstT + (size_t)bh * GAT_N;
  const v16h*  hp    = ((const v16h*)hB) + (size_t)bh * 4 * (GAT_N / 32) * 32;

  v8f a00 = {}, a01 = {}, a02 = {}, a03 = {};
  v8f a10 = {}, a11 = {}, a12 = {}, a13 = {};
  float z0 = 0.f, z1 = 0.f;

  for (int mc = 0; mc < GAT_N / 32; ++mc) {
    const int m0 = mc * 32;
    const float4* dq  = (const float4*)(dRow + m0 + koff);
    const float4* aq0 = (const float4*)(adjR0 + m0 + koff);
    const float4* aq1 = (const float4*)(adjR1 + m0 + koff);
    float db[16], ab0[16], ab1[16];
    ((float4*)db)[0]  = dq[0];  ((float4*)db)[1]  = dq[1];
    ((float4*)db)[2]  = dq[4];  ((float4*)db)[3]  = dq[5];
    ((float4*)ab0)[0] = aq0[0]; ((float4*)ab0)[1] = aq0[1];
    ((float4*)ab0)[2] = aq0[4]; ((float4*)ab0)[3] = aq0[5];
    ((float4*)ab1)[0] = aq1[0]; ((float4*)ab1)[1] = aq1[1];
    ((float4*)ab1)[2] = aq1[4]; ((float4*)ab1)[3] = aq1[5];

    if (mc + 2 < GAT_N / 32) {
      __builtin_prefetch(adjR0 + m0 + 64, 0, 0);
      __builtin_prefetch(adjR1 + m0 + 64, 0, 0);
    }

    v16h av0, av1;
#pragma unroll
    for (int j = 0; j < 16; ++j) {
      const float dv = db[j];
      float e0 = gat_lrelu(c0 + dv);
      float e1 = gat_lrelu(c1 + dv);
      float p0 = (ab0[j] > 0.f) ? __expf(e0 - sh0) : 0.f;
      float p1 = (ab1[j] > 0.f) ? __expf(e1 - sh1) : 0.f;
      z0 += p0;
      z1 += p1;
      av0[j] = (_Float16)p0;
      av1[j] = (_Float16)p1;
    }

    const v16h* hpc = hp + (size_t)mc * 32 + lane;
    if (mc + 1 < GAT_N / 32) {
      __builtin_prefetch(hpc + 32 + 0 * (GAT_N / 32) * 32, 0, 0);
      __builtin_prefetch(hpc + 32 + 1 * (GAT_N / 32) * 32, 0, 0);
      __builtin_prefetch(hpc + 32 + 2 * (GAT_N / 32) * 32, 0, 0);
      __builtin_prefetch(hpc + 32 + 3 * (GAT_N / 32) * 32, 0, 0);
    }
    v16h b0 = hpc[0 * (GAT_N / 32) * 32];
    v16h b1 = hpc[1 * (GAT_N / 32) * 32];
    v16h b2 = hpc[2 * (GAT_N / 32) * 32];
    v16h b3 = hpc[3 * (GAT_N / 32) * 32];
    a00 = __builtin_amdgcn_wmma_f32_16x16x32_f16(false, av0, false, b0, (short)0, a00, false, false);
    a10 = __builtin_amdgcn_wmma_f32_16x16x32_f16(false, av1, false, b0, (short)0, a10, false, false);
    a01 = __builtin_amdgcn_wmma_f32_16x16x32_f16(false, av0, false, b1, (short)0, a01, false, false);
    a11 = __builtin_amdgcn_wmma_f32_16x16x32_f16(false, av1, false, b1, (short)0, a11, false, false);
    a02 = __builtin_amdgcn_wmma_f32_16x16x32_f16(false, av0, false, b2, (short)0, a02, false, false);
    a12 = __builtin_amdgcn_wmma_f32_16x16x32_f16(false, av1, false, b2, (short)0, a12, false, false);
    a03 = __builtin_amdgcn_wmma_f32_16x16x32_f16(false, av0, false, b3, (short)0, a03, false, false);
    a13 = __builtin_amdgcn_wmma_f32_16x16x32_f16(false, av1, false, b3, (short)0, a13, false, false);
  }

  // fold per-lane partial row sums: lane L holds partial for row L%16
  float zf0 = z0 + __shfl_xor(z0, 16, 32);
  float zf1 = z1 + __shfl_xor(z1, 16, 32);
  const int rowSel = (lane & 16) >> 1;  // 8 * (lane>=16), per C-matrix layout
  float* orow0 = out + ((size_t)b * GAT_N + n0) * GAT_HF + hh * GAT_F + r;
  float* orow1 = orow0 + (size_t)16 * GAT_HF;
#pragma unroll
  for (int j = 0; j < 8; ++j) {
    const int row = rowSel + j;
    const float inv0 = __builtin_amdgcn_rcpf(__shfl(zf0, row, 32));
    const float inv1 = __builtin_amdgcn_rcpf(__shfl(zf1, row, 32));
    const size_t rb = (size_t)row * GAT_HF;
    orow0[rb +  0] = a00[j] * inv0;
    orow0[rb + 16] = a01[j] * inv0;
    orow0[rb + 32] = a02[j] * inv0;
    orow0[rb + 48] = a03[j] * inv0;
    orow1[rb +  0] = a10[j] * inv1;
    orow1[rb + 16] = a11[j] * inv1;
    orow1[rb + 32] = a12[j] * inv1;
    orow1[rb + 48] = a13[j] * inv1;
  }
}

// ---------------------------------------------------------------------------
extern "C" void kernel_launch(void* const* d_in, const int* in_sizes, int n_in,
                              void* d_out, int out_size, void* d_ws, size_t ws_size,
                              hipStream_t stream) {
  const float* x    = (const float*)d_in[0];   // [B,N,256]
  const float* adj  = (const float*)d_in[1];   // [N,N]
  const float* W    = (const float*)d_in[2];   // [256,256]
  const float* aSrc = (const float*)d_in[3];   // [H,F]
  const float* aDst = (const float*)d_in[4];   // [H,F]
  float* out = (float*)d_out;                  // [B,N,256]

  char* ws = (char*)d_ws;
  float*    h_f32 = (float*)ws;                       // 8 MB
  _Float16* hB    = (_Float16*)(ws + 8388608);        // 4 MB (swizzled f16 h)
  float*    sSrcT = (float*)(ws + 12582912);          // 128 KB [B,H,N]
  float*    sDstT = (float*)(ws + 12713984);          // 128 KB [B,H,N]
  float*    dmax  = (float*)(ws + 12845056);          // 32 B   [B,H] (+pad)
  _Float16* WhiB  = (_Float16*)(ws + 12845312);       // 128 KB swizzled W hi
  _Float16* WloB  = (_Float16*)(ws + 12976384);       // 128 KB swizzled W lo

  k0_wprep<<<(GAT_HF * GAT_INF) / 256, 256, 0, stream>>>(W, WhiB, WloB);
  k1_wmma<<<(GAT_B * GAT_N) / 32, 256, 0, stream>>>(x, WhiB, WloB, h_f32, hB);
  k1b_scores<<<(GAT_B * GAT_N * GAT_H) / 256, 256, 0, stream>>>(h_f32, aSrc, aDst, sSrcT, sDstT);
  k1c_dmax<<<GAT_B * GAT_H, 256, 0, stream>>>(sDstT, dmax);
  k2_attn<<<dim3(GAT_N / 32, GAT_B), 128, 0, stream>>>(adj, sSrcT, sDstT, dmax, hB, out);
}
